// Model_38963943309578
// MI455X (gfx1250) — matrix operand order repaired and checked by reading, not compiled, not run on hardware
//
#include <hip/hip_runtime.h>
#include <hip/hip_bf16.h>
#include <math.h>

// ---------------------------------------------------------------- constants
#define BB   32
#define TT   2048
#define T2   1024
#define CC   64
#define FIN  8
#define CF   32
#define HID  128
#define NCLS 4

typedef _Float16 v16h __attribute__((ext_vector_type(16)));
typedef float    v8f  __attribute__((ext_vector_type(8)));

union Frag { v16h h; unsigned u[8]; };

// ---------------------------------------------------------------- k0: mix_w -> f16 in WMMA-fragment-major layout
// layout: [ktile 0..63][ntile 0..7][lane 0..31][16 halfs]; chunk half s is K = ktile*32 + (s>=8?16:0) + (lane>>4)*8 + (s&7) at N = ntile*16 + (lane&15)
__global__ __launch_bounds__(256) void k0_cvt(const float* __restrict__ mix_w,
                                              _Float16* __restrict__ mixw_t,
                                              float* __restrict__ spk_accum) {
    int i = blockIdx.x * 256 + threadIdx.x;          // 0..262143 = output index
    int j2   = i & 15;
    int lane = (i >> 4) & 31;
    int ni   = (i >> 9) & 7;
    int kt   = i >> 12;
    mixw_t[i] = (_Float16)mix_w[(size_t)(kt * 32 + ((j2 & 8) ? 16 : 0) + (lane >> 4) * 8 + (j2 & 7)) * HID + ni * 16 + (lane & 15)];
    if (i == 0) *spk_accum = 0.0f;
}

// ---------------------------------------------------------------- k1: conv(5x1,s2) + bias + relu + LN1 -> Xc (f16)
__global__ __launch_bounds__(256) void k1_conv_ln(const float* __restrict__ X,
                                                  const float* __restrict__ conv_w,
                                                  const float* __restrict__ conv_b,
                                                  const float* __restrict__ ln1_s,
                                                  const float* __restrict__ ln1_b,
                                                  _Float16* __restrict__ Xc_h) {
    __shared__ float sw[5 * 8 * 32];
    __shared__ float sb[32], sg[32], sbt[32];
    for (int i = threadIdx.x; i < 1280; i += 256) sw[i] = conv_w[i];
    if (threadIdx.x < 32) {
        sb[threadIdx.x]  = conv_b[threadIdx.x];
        sg[threadIdx.x]  = ln1_s[threadIdx.x];
        sbt[threadIdx.x] = ln1_b[threadIdx.x];
    }
    __syncthreads();

    int gid = blockIdx.x * 256 + threadIdx.x;        // b*65536 + t2*64 + c
    int b   = gid >> 16;
    int rem = gid & 65535;
    int t2  = rem >> 6;
    int c   = rem & 63;

    float acc[CF];
#pragma unroll
    for (int f = 0; f < CF; ++f) acc[f] = sb[f];

#pragma unroll
    for (int kt = 0; kt < 5; ++kt) {
        int tin = 2 * t2 - 1 + kt;                   // SAME pad: lo=1
        if (tin < 0 || tin >= TT) continue;
        const float* xp = X + (((size_t)b * TT + tin) * CC + c) * FIN;
        float xv[FIN];
#pragma unroll
        for (int ci = 0; ci < FIN; ++ci) xv[ci] = xp[ci];
#pragma unroll
        for (int ci = 0; ci < FIN; ++ci) {
            const float* wrow = &sw[(kt * FIN + ci) * CF];
#pragma unroll
            for (int f = 0; f < CF; ++f) acc[f] = fmaf(xv[ci], wrow[f], acc[f]);
        }
    }
    float mu = 0.0f;
#pragma unroll
    for (int f = 0; f < CF; ++f) { acc[f] = fmaxf(acc[f], 0.0f); mu += acc[f]; }
    mu *= (1.0f / CF);
    float var = 0.0f;
#pragma unroll
    for (int f = 0; f < CF; ++f) { float d = acc[f] - mu; var += d * d; }
    var *= (1.0f / CF);
    float rstd = rsqrtf(var + 1e-6f);

    _Float16* dst = Xc_h + (size_t)gid * CF;
#pragma unroll
    for (int g = 0; g < 4; ++g) {
        union { _Float16 h[8]; uint4 v; } pk;
#pragma unroll
        for (int j = 0; j < 8; ++j) {
            int f = g * 8 + j;
            pk.h[j] = (_Float16)((acc[f] - mu) * rstd * sg[f] + sbt[f]);
        }
        ((uint4*)dst)[g] = pk.v;
    }
}

// ---------------------------------------------------------------- k2: time-mean -> x_spat (B,C,32) and chan_mean (B,C)
__global__ __launch_bounds__(256) void k2_means(const _Float16* __restrict__ Xc_h,
                                                float* __restrict__ x_spat,
                                                float* __restrict__ chan_mean) {
    int bc = blockIdx.x;                             // b*64 + c
    int b = bc >> 6, c = bc & 63;
    int f = threadIdx.x & 31, tt = threadIdx.x >> 5;
    const _Float16* base = Xc_h + (((size_t)b * T2) * CC + c) * CF + f;
    float acc = 0.0f;
    for (int t = tt; t < T2; t += 8) acc += (float)base[(size_t)t * (CC * CF)];
    __shared__ float sred[8][32];
    sred[tt][f] = acc;
    __syncthreads();
    if (tt == 0) {
        float s = 0.0f;
#pragma unroll
        for (int i = 0; i < 8; ++i) s += sred[i][f];
        s *= (1.0f / T2);
        x_spat[(size_t)bc * CF + f] = s;
        sred[0][f] = s;
    }
    __syncthreads();
    if (threadIdx.x == 0) {
        float s = 0.0f;
#pragma unroll
        for (int i = 0; i < 32; ++i) s += sred[0][i];
        chan_mean[bc] = s * (1.0f / CF);
    }
}

// ---------------------------------------------------------------- k3: q/k proj + softmax adj (f16) + channel gate ca
__global__ __launch_bounds__(64) void k3_attn(const float* __restrict__ x_spat,
                                              const float* __restrict__ chan_mean,
                                              const float* __restrict__ q_w, const float* __restrict__ q_b,
                                              const float* __restrict__ k_w, const float* __restrict__ k_b,
                                              const float* __restrict__ ca_w, const float* __restrict__ ca_b,
                                              _Float16* __restrict__ adj_h, float* __restrict__ ca) {
    int b = blockIdx.x, c = threadIdx.x;
    __shared__ float sq[CC][32], sk[CC][32];
    float xs[32];
#pragma unroll
    for (int f = 0; f < 32; ++f) xs[f] = x_spat[((size_t)b * CC + c) * CF + f];
#pragma unroll 4
    for (int a = 0; a < 32; ++a) {
        float qa = q_b[a], ka = k_b[a];
#pragma unroll
        for (int f = 0; f < 32; ++f) {
            qa = fmaf(xs[f], q_w[f * 32 + a], qa);
            ka = fmaf(xs[f], k_w[f * 32 + a], ka);
        }
        sq[c][a] = qa; sk[c][a] = ka;
    }
    __syncthreads();
    float sc[CC];
    float mx = -3.4e38f;
#pragma unroll 8
    for (int d = 0; d < CC; ++d) {
        float s = 0.0f;
#pragma unroll
        for (int a = 0; a < 32; ++a) s = fmaf(sq[c][a], sk[d][a], s);
        s *= 0.17677669529663687f;                    // 1/sqrt(32)
        sc[d] = s;
        mx = fmaxf(mx, s);
    }
    float sum = 0.0f;
#pragma unroll 8
    for (int d = 0; d < CC; ++d) { float e = __expf(sc[d] - mx); sc[d] = e; sum += e; }
    float inv = 1.0f / sum;
    _Float16* arow = adj_h + ((size_t)b * CC + c) * CC;
#pragma unroll 8
    for (int d = 0; d < CC; ++d) arow[d] = (_Float16)(sc[d] * inv);
    float a = ca_b[c];
    for (int d = 0; d < CC; ++d) a = fmaf(chan_mean[b * CC + d], ca_w[d * CC + c], a);
    ca[b * CC + c] = 1.0f / (1.0f + __expf(-a));
}

// ---------------------------------------------------------------- k45: FUSED graph-GEMM + gate + residual + mixer + LN2
// block = 16 t-rows (one b). Phase p covers c-rows [p*32,p*32+32) == mixer K-half [p*1024, p*1024+1024).
// Graph phase: wave handles 2 t's, WMMA Xg = adj @ Xc, epilogue Xr = Xg*ca + Xc written straight into LDS (f16).
// Mixer phase: wave = N-tile of HID, A-fragments from LDS, B-fragments from tiled mix_w; accumulator spans phases.
__global__ __launch_bounds__(256) void k45_fused(const _Float16* __restrict__ Xc_h,
                                                 const _Float16* __restrict__ adj_h,
                                                 const float* __restrict__ ca,
                                                 const _Float16* __restrict__ mixw_t,
                                                 const float* __restrict__ mix_b,
                                                 const float* __restrict__ ln2_s,
                                                 const float* __restrict__ ln2_b,
                                                 float* __restrict__ Xm) {
    __shared__ __align__(16) _Float16 sA[16 * 1024];  // 32 KB: 16 t-rows x current K-half
    __shared__ float sXm[16 * HID];                   // 8 KB
    __shared__ float sMu[16], sRs[16];
    int lane = threadIdx.x & 31, wave = threadIdx.x >> 5;
    int row_base = blockIdx.x * 16;                   // global bt base; T2%16==0 -> single b per block
    int b = row_base >> 10;
    int mrow = lane & 15, hb = lane >> 4;

    // --- load B-fragments of Xc for this wave's two t's, once (held in VGPRs across phases)
    Frag bf[2][2][2];                                 // [subT][ki][ni]; lane = N, VGPR j = K pair
#pragma unroll
    for (int subT = 0; subT < 2; ++subT) {
        const _Float16* xc = Xc_h + (size_t)(row_base + wave * 2 + subT) * (CC * CF);
#pragma unroll
        for (int ki = 0; ki < 2; ++ki)
#pragma unroll
            for (int ni = 0; ni < 2; ++ni) {
                const _Float16* p = xc + (ki * 32) * CF + ni * 16 + mrow;
#pragma unroll
                for (int j = 0; j < 8; ++j) {
                    int kb = (j & 3) * 2 + ((j >= 4) ? 16 : 0) + hb * 8;
                    bf[subT][ki][ni].h[2 * j]     = p[kb * CF];
                    bf[subT][ki][ni].h[2 * j + 1] = p[(kb + 1) * CF];
                }
            }
    }
    const _Float16* adjb = adj_h + (size_t)b * (CC * CC);
    const float*    cab  = ca + b * CC;

    v8f macc = {};                                    // mixer accumulator, carried across both phases
#pragma unroll
    for (int p = 0; p < 2; ++p) {
        __syncthreads();                              // prev phase's sA fully consumed before overwrite
        // ---- graph GEMM half: c-rows mi in {2p, 2p+1}
#pragma unroll
        for (int subT = 0; subT < 2; ++subT) {
            int tloc = wave * 2 + subT;
            const _Float16* xc = Xc_h + (size_t)(row_base + tloc) * (CC * CF);
#pragma unroll
            for (int mi2 = 0; mi2 < 2; ++mi2) {
                int mi = p * 2 + mi2;
                Frag af[2];
#pragma unroll
                for (int ki = 0; ki < 2; ++ki) {
                    int m = mi * 16 + mrow;
#pragma unroll
                    for (int j = 0; j < 8; ++j) {
                        int kb = ki * 32 + (j & 3) * 2 + ((j >= 4) ? 16 : 0) + hb * 8;
                        af[ki].u[j] = *(const unsigned*)(adjb + m * CC + kb);
                    }
                }
#pragma unroll
                for (int ni = 0; ni < 2; ++ni) {
                    v8f cacc = {};
                    cacc = __builtin_amdgcn_wmma_f32_16x16x32_f16(false, af[0].h, false, bf[subT][0][ni].h,
                                                                  (short)0, cacc, false, false);
                    cacc = __builtin_amdgcn_wmma_f32_16x16x32_f16(false, af[1].h, false, bf[subT][1][ni].h,
                                                                  (short)0, cacc, false, false);
                    int fcol = ni * 16 + mrow;
#pragma unroll
                    for (int r = 0; r < 8; ++r) {
                        int crow = mi * 16 + r + hb * 8;
                        float v = fmaf(cacc[r], cab[crow], (float)xc[crow * CF + fcol]);
                        sA[tloc * 1024 + (crow - p * 32) * CF + fcol] = (_Float16)v;  // Xr -> LDS only
                    }
                }
            }
        }
        __syncthreads();
        // ---- mixer over this K-half: 32 WMMA steps; A from LDS, B from fragment-major mix_w
        for (int ki = 0; ki < 32; ++ki) {
            Frag a, bfr;
#pragma unroll
            for (int j = 0; j < 8; ++j) {
                int kb = ki * 32 + (j & 3) * 2 + ((j >= 4) ? 16 : 0) + hb * 8;
                a.u[j] = *(const unsigned*)(&sA[mrow * 1024 + kb]);
            }
            const unsigned* bp =
                (const unsigned*)(mixw_t + ((size_t)(((p * 32 + ki) * 8 + wave) * 32 + lane)) * 16);
#pragma unroll
            for (int j = 0; j < 8; ++j) bfr.u[j] = bp[j];
            macc = __builtin_amdgcn_wmma_f32_16x16x32_f16(false, a.h, false, bfr.h,
                                                          (short)0, macc, false, false);
        }
    }
    // ---- mixer epilogue: bias + relu (C/D layout: M = r + 8*hi, N = lane&15)
    {
        int hcol = wave * 16 + mrow;
        float bh = mix_b[hcol];
#pragma unroll
        for (int r = 0; r < 8; ++r)
            sXm[(r + hb * 8) * HID + hcol] = fmaxf(macc[r] + bh, 0.0f);
    }
    __syncthreads();
    if (threadIdx.x < 16) {
        int row = threadIdx.x;
        float mu = 0.0f;
        for (int h = 0; h < HID; ++h) mu += sXm[row * HID + h];
        mu *= (1.0f / HID);
        float var = 0.0f;
        for (int h = 0; h < HID; ++h) { float d = sXm[row * HID + h] - mu; var += d * d; }
        var *= (1.0f / HID);
        sMu[row] = mu;
        sRs[row] = rsqrtf(var + 1e-6f);
    }
    __syncthreads();
#pragma unroll
    for (int i = 0; i < 8; ++i) {
        int idx = threadIdx.x + i * 256;
        int row = idx >> 7, h = idx & 127;
        Xm[((size_t)row_base + row) * HID + h] =
            (sXm[idx] - sMu[row]) * sRs[row] * ln2_s[h] + ln2_b[h];
    }
}

// ---------------------------------------------------------------- k6: LIF scan over t; feat = mean_t spk; spike total
__global__ __launch_bounds__(256) void k6_lif(const float* __restrict__ Xm,
                                              const float* __restrict__ decay_logit,
                                              float* __restrict__ feat,
                                              float* __restrict__ spk_accum) {
    int gid = blockIdx.x * 256 + threadIdx.x;        // (b,h)
    int b = gid >> 7, h = gid & 127;
    float decay = 1.0f / (1.0f + __expf(-decay_logit[h]));
    float mem = 0.0f, cnt = 0.0f;
    const float* xp = Xm + (size_t)b * T2 * HID + h;
    for (int t = 0; t < T2; ++t) {
        mem = fmaf(mem, decay, xp[(size_t)t * HID]);
        float spk = (mem > 0.5f) ? 1.0f : 0.0f;
        cnt += spk;
        mem -= spk * 0.5f;
    }
    feat[gid] = cnt * (1.0f / T2);
    __shared__ float sred[256];
    sred[threadIdx.x] = cnt;
    __syncthreads();
    for (int s = 128; s > 0; s >>= 1) {
        if (threadIdx.x < s) sred[threadIdx.x] += sred[threadIdx.x + s];
        __syncthreads();
    }
    if (threadIdx.x == 0) atomicAdd(spk_accum, sred[0]);
}

// ---------------------------------------------------------------- k7: head
__global__ __launch_bounds__(128) void k7_head(const float* __restrict__ feat,
                                               const float* __restrict__ fc_w,
                                               const float* __restrict__ fc_b,
                                               const float* __restrict__ out_w,
                                               const float* __restrict__ out_b,
                                               const float* __restrict__ spk_accum,
                                               float* __restrict__ out) {
    __shared__ float sf[BB * HID];
    __shared__ float sfc[BB * HID];
    for (int i = threadIdx.x; i < BB * HID; i += 128) sf[i] = feat[i];
    __syncthreads();
    int j = threadIdx.x;
    float bj = fc_b[j];
    for (int b = 0; b < BB; ++b) {
        float acc = bj;
        for (int h = 0; h < HID; ++h) acc = fmaf(sf[b * HID + h], fc_w[h * HID + j], acc);
        sfc[b * HID + j] = fmaxf(acc, 0.0f);
    }
    __syncthreads();
    int b = threadIdx.x >> 2, n = threadIdx.x & 3;
    float acc = out_b[n];
    for (int jj = 0; jj < HID; ++jj) acc = fmaf(sfc[b * HID + jj], out_w[jj * NCLS + n], acc);
    out[threadIdx.x] = acc;
    if (threadIdx.x == 0)
        out[BB * NCLS] = spk_accum[0] * (1.0f / ((float)BB * T2 * HID));
}

// ---------------------------------------------------------------- launch
extern "C" void kernel_launch(void* const* d_in, const int* in_sizes, int n_in,
                              void* d_out, int out_size, void* d_ws, size_t ws_size,
                              hipStream_t stream) {
    const float* X           = (const float*)d_in[0];
    const float* conv_w      = (const float*)d_in[1];
    const float* conv_b      = (const float*)d_in[2];
    const float* ln1_s       = (const float*)d_in[3];
    const float* ln1_b       = (const float*)d_in[4];
    const float* q_w         = (const float*)d_in[5];
    const float* q_b         = (const float*)d_in[6];
    const float* k_w         = (const float*)d_in[7];
    const float* k_b         = (const float*)d_in[8];
    const float* ca_w        = (const float*)d_in[9];
    const float* ca_b        = (const float*)d_in[10];
    const float* mix_w       = (const float*)d_in[11];
    const float* mix_b       = (const float*)d_in[12];
    const float* ln2_s       = (const float*)d_in[13];
    const float* ln2_b       = (const float*)d_in[14];
    const float* decay_logit = (const float*)d_in[15];
    const float* fc_w        = (const float*)d_in[16];
    const float* fc_b        = (const float*)d_in[17];
    const float* out_w       = (const float*)d_in[18];
    const float* out_b       = (const float*)d_in[19];
    float* out = (float*)d_out;

    char* ws = (char*)d_ws;
    const size_t XC_BYTES   = (size_t)BB * T2 * CC * CF * 2;   // 134 MB
    const size_t XM_BYTES   = (size_t)BB * T2 * HID * 4;       // 16.8 MB
    const size_t MIXW_BYTES = (size_t)CC * CF * HID * 2;       // 512 KB
    const size_t XSP_BYTES  = (size_t)BB * CC * CF * 4;
    const size_t ADJ_BYTES  = (size_t)BB * CC * CC * 2;
    const size_t CHN_BYTES  = (size_t)BB * CC * 4;
    const size_t CA_BYTES   = (size_t)BB * CC * 4;
    const size_t FEAT_BYTES = (size_t)BB * HID * 4;

    _Float16* Xc_h      = (_Float16*)ws;               ws += XC_BYTES;
    float*    Xm        = (float*)ws;                  ws += XM_BYTES;
    _Float16* mixw_t    = (_Float16*)ws;               ws += MIXW_BYTES;
    float*    x_spat    = (float*)ws;                  ws += XSP_BYTES;
    _Float16* adj_h     = (_Float16*)ws;               ws += ADJ_BYTES;
    float*    chan_mean = (float*)ws;                  ws += CHN_BYTES;
    float*    ca        = (float*)ws;                  ws += CA_BYTES;
    float*    feat      = (float*)ws;                  ws += FEAT_BYTES;
    float*    spk_accum = (float*)ws;                  ws += 256;

    k0_cvt<<<(CC * CF * HID) / 256, 256, 0, stream>>>(mix_w, mixw_t, spk_accum);
    k1_conv_ln<<<(BB * T2 * CC) / 256, 256, 0, stream>>>(X, conv_w, conv_b, ln1_s, ln1_b, Xc_h);
    k2_means<<<BB * CC, 256, 0, stream>>>(Xc_h, x_spat, chan_mean);
    k3_attn<<<BB, 64, 0, stream>>>(x_spat, chan_mean, q_w, q_b, k_w, k_b, ca_w, ca_b, adj_h, ca);
    k45_fused<<<(BB * T2) / 16, 256, 0, stream>>>(Xc_h, adj_h, ca, mixw_t, mix_b, ln2_s, ln2_b, Xm);
    k6_lif<<<(BB * HID) / 256, 256, 0, stream>>>(Xm, decay_logit, feat, spk_accum);
    k7_head<<<1, 128, 0, stream>>>(feat, fc_w, fc_b, out_w, out_b, spk_accum, out);
}
